// EnhancedGATLSTMWithAttention_24429773980443
// MI455X (gfx1250) — matrix operand order repaired and checked by reading, not compiled, not run on hardware
//
#include <hip/hip_runtime.h>
#include <hip/hip_bf16.h>

typedef __attribute__((ext_vector_type(16))) _Float16 v16h;
typedef __attribute__((ext_vector_type(8)))  float    v8f;

#define NNODES 50000
#define NEDGES 800000
#define ETOT   (NEDGES + NNODES)
#define NGRAPH 64
#define DH     128
#define LSTMH  256
#define EPSN   1e-5f
#define LCHUNK 8192

// ---------------- helpers ----------------
__device__ __forceinline__ unsigned f2ord(float f) {
  unsigned u = __float_as_uint(f);
  return (u & 0x80000000u) ? ~u : (u | 0x80000000u);
}
__device__ __forceinline__ float ord2f(unsigned o) {
  unsigned u = (o & 0x80000000u) ? (o & 0x7FFFFFFFu) : ~o;
  return __uint_as_float(u);
}
__device__ __forceinline__ float sigm(float x) { return 1.0f / (1.0f + expf(-x)); }

__global__ void k_zero(float* p, long n) {
  long i = (long)blockIdx.x * 256 + threadIdx.x;
  if (i < n) p[i] = 0.0f;
}

__global__ void k_count(const int* __restrict__ bid, float* cnt) {
  int n = blockIdx.x * 256 + threadIdx.x;
  if (n < NNODES) atomicAdd(&cnt[bid[n]], 1.0f);
}

// ------------- input projection: feat[N,33] @ W_in[33,128] + b_in -------------
__global__ __launch_bounds__(128)
void k_inproj(const int* __restrict__ lab, const int* __restrict__ typ,
              const float* __restrict__ nf, const float* __restrict__ embL,
              const float* __restrict__ embT, const float* __restrict__ Win,
              const float* __restrict__ bin, float* __restrict__ x) {
  int n = blockIdx.x, d = threadIdx.x;
  __shared__ float f[33];
  if (d < 16)       f[d] = embL[(size_t)lab[n] * 16 + d];
  else if (d < 32)  f[d] = embT[(size_t)typ[n] * 16 + (d - 16)];
  else if (d == 32) f[32] = nf[n];
  __syncthreads();
  float s = bin[d];
#pragma unroll
  for (int k = 0; k < 33; ++k) s += f[k] * Win[k * DH + d];
  x[(size_t)n * DH + d] = s;
}

// ------------- operand packing into WMMA fragment order (f32 -> f16) -------------
// A fragment layout (ISA 7.12.2): Ap[((mt*KT + kt)*32 + lane)*16 + i]
//   m = mt*16 + lane%16 ; k = kt*32 + (lane/16)*8 + (i%8) + (i/8)*16
__global__ void k_packA(const float* __restrict__ A, _Float16* __restrict__ Ap,
                        int Mrows, int K) {
  int KT = K >> 5;
  long idx = (long)blockIdx.x * 256 + threadIdx.x;
  long total = (long)((Mrows + 15) >> 4) * KT * 512;
  if (idx >= total) return;
  int i    = (int)(idx & 15);
  int lane = (int)((idx >> 4) & 31);
  long t   = idx >> 9;
  int kt   = (int)(t % KT);
  int mt   = (int)(t / KT);
  int m = mt * 16 + (lane & 15);
  int k = kt * 32 + (lane >> 4) * 8 + (i & 7) + (i >> 3) * 16;
  Ap[idx] = (m < Mrows) ? (_Float16)A[(size_t)m * K + k] : (_Float16)0.0f;
}

// B fragment layout: Bp[((nt*KT + kt)*32 + lane)*16 + i]
//   n = nt*16 + lane%16 ; k = kt*32 + (lane/16)*8 + (i%8) + (i/8)*16 ; B(k,n) = W[k*sWk + n*sWn]
__global__ void k_packB(const float* __restrict__ W, int sWk, int sWn,
                        _Float16* __restrict__ Bp, int K, int Nc) {
  int KT = K >> 5;
  long idx = (long)blockIdx.x * 256 + threadIdx.x;
  long total = (long)(Nc >> 4) * KT * 512;
  if (idx >= total) return;
  int i    = (int)(idx & 15);
  int lane = (int)((idx >> 4) & 31);
  long t   = idx >> 9;
  int kt   = (int)(t % KT);
  int nt   = (int)(t / KT);
  int n = nt * 16 + (lane & 15);
  int k = kt * 32 + (lane >> 4) * 8 + (i & 7) + (i >> 3) * 16;
  Bp[idx] = (_Float16)W[(size_t)k * sWk + (size_t)n * sWn];
}

// ------------- WMMA GEMM on packed fragments -------------
// wave w of block computes m-tile mt = mt0 + blockIdx.x*4 + w, n-tiles blockIdx.y*4 .. +3
__global__ __launch_bounds__(128)
void k_gemm_wmma(const _Float16* __restrict__ Ap, const _Float16* __restrict__ Bp,
                 const float* __restrict__ pb1, const float* __restrict__ pb2,
                 float* __restrict__ C, int ldc, int Mrows, int KT,
                 int mt0, int rowbase) {
  const int wave = threadIdx.x >> 5;
  const int lane = threadIdx.x & 31;
  const int mloc = lane & 15;
  const int mid  = lane >> 4;
  const int mt   = mt0 + blockIdx.x * 4 + wave;
  const int MT   = (Mrows + 15) >> 4;
  if (mt >= MT) return;  // wave-uniform
  const int ntg0 = blockIdx.y * 4;

  const _Float16* Aq  = Ap + ((size_t)mt * KT * 32 + lane) * 16;
  const _Float16* Bq0 = Bp + ((size_t)(ntg0 + 0) * KT * 32 + lane) * 16;
  const _Float16* Bq1 = Bp + ((size_t)(ntg0 + 1) * KT * 32 + lane) * 16;
  const _Float16* Bq2 = Bp + ((size_t)(ntg0 + 2) * KT * 32 + lane) * 16;
  const _Float16* Bq3 = Bp + ((size_t)(ntg0 + 3) * KT * 32 + lane) * 16;

  v8f acc0 = {0.f,0.f,0.f,0.f,0.f,0.f,0.f,0.f};
  v8f acc1 = acc0, acc2 = acc0, acc3 = acc0;

  for (int kt = 0; kt < KT; ++kt) {
    v16h a  = *(const v16h*)Aq;
    v16h b0 = *(const v16h*)Bq0;
    v16h b1 = *(const v16h*)Bq1;
    v16h b2 = *(const v16h*)Bq2;
    v16h b3 = *(const v16h*)Bq3;
    acc0 = __builtin_amdgcn_wmma_f32_16x16x32_f16(false, a, false, b0, (short)0, acc0, false, false);
    acc1 = __builtin_amdgcn_wmma_f32_16x16x32_f16(false, a, false, b1, (short)0, acc1, false, false);
    acc2 = __builtin_amdgcn_wmma_f32_16x16x32_f16(false, a, false, b2, (short)0, acc2, false, false);
    acc3 = __builtin_amdgcn_wmma_f32_16x16x32_f16(false, a, false, b3, (short)0, acc3, false, false);
    Aq += 512; Bq0 += 512; Bq1 += 512; Bq2 += 512; Bq3 += 512;
  }
  // C/D layout: n = (ntg0+nt)*16 + lane%16, m = mt*16 + (lane/16)*8 + r
#define STORE_TILE(nt, accv)                                                           \
  {                                                                                    \
    const int n = (ntg0 + (nt)) * 16 + mloc;                                           \
    const float bb = (pb1 ? pb1[n] : 0.0f) + (pb2 ? pb2[n] : 0.0f);                    \
    _Pragma("unroll")                                                                  \
    for (int r = 0; r < 8; ++r) {                                                      \
      int m = mt * 16 + mid * 8 + r;                                                   \
      if (m < Mrows) C[(size_t)(m - rowbase) * ldc + n] = accv[r] + bb;                \
    }                                                                                  \
  }
  STORE_TILE(0, acc0) STORE_TILE(1, acc1) STORE_TILE(2, acc2) STORE_TILE(3, acc3)
#undef STORE_TILE
}

// ------------- GAT edge kernels (wave32 per edge, float4 per lane) -------------
__global__ __launch_bounds__(256)
void k_score(const float* __restrict__ xl, const float* __restrict__ xr,
             const int* __restrict__ eidx, const float* __restrict__ att,
             float* __restrict__ score, unsigned* __restrict__ smax) {
  int wave = threadIdx.x >> 5, lane = threadIdx.x & 31;
  long e = (long)blockIdx.x * 8 + wave;
  if (e >= ETOT) return;
  int src, dst;
  if (e < NEDGES) { src = eidx[e]; dst = eidx[NEDGES + e]; }
  else            { src = dst = (int)(e - NEDGES); }
  float4 xlv = ((const float4*)xl)[(size_t)src * 32 + lane];
  float4 xrv = ((const float4*)xr)[(size_t)dst * 32 + lane];
  float4 av  = ((const float4*)att)[lane];
  float s = 0.0f, v;
  v = xrv.x + xlv.x; s += (v > 0.0f ? v : 0.2f * v) * av.x;
  v = xrv.y + xlv.y; s += (v > 0.0f ? v : 0.2f * v) * av.y;
  v = xrv.z + xlv.z; s += (v > 0.0f ? v : 0.2f * v) * av.z;
  v = xrv.w + xlv.w; s += (v > 0.0f ? v : 0.2f * v) * av.w;
  s += __shfl_xor(s, 1, 32);
  s += __shfl_xor(s, 2, 32);
  s += __shfl_xor(s, 4, 32);  // 8 lanes per head reduced
  if ((lane & 7) == 0) {
    int h = lane >> 3;
    score[e * 4 + h] = s;
    atomicMax(&smax[(size_t)dst * 4 + h], f2ord(s));
  }
}

__global__ void k_expdenom(const int* __restrict__ eidx, const unsigned* __restrict__ smax,
                           float* __restrict__ score, float* __restrict__ denom) {
  long idx = (long)blockIdx.x * 256 + threadIdx.x;
  if (idx >= (long)ETOT * 4) return;
  long e = idx >> 2; int h = (int)(idx & 3);
  int dst = (e < NEDGES) ? eidx[NEDGES + e] : (int)(e - NEDGES);
  float m = ord2f(smax[(size_t)dst * 4 + h]);
  float a = expf(score[idx] - m);
  score[idx] = a;
  atomicAdd(&denom[(size_t)dst * 4 + h], a);
}

__global__ __launch_bounds__(256)
void k_aggregate(const float* __restrict__ xl, const int* __restrict__ eidx,
                 const float* __restrict__ score, const float* __restrict__ denom,
                 float* __restrict__ agg) {
  int wave = threadIdx.x >> 5, lane = threadIdx.x & 31;
  long e = (long)blockIdx.x * 8 + wave;
  if (e >= ETOT) return;
  int src, dst;
  if (e < NEDGES) { src = eidx[e]; dst = eidx[NEDGES + e]; }
  else            { src = dst = (int)(e - NEDGES); }
  int h = lane >> 3;
  float alpha = score[e * 4 + h] / denom[(size_t)dst * 4 + h];
  float4 xlv = ((const float4*)xl)[(size_t)src * 32 + lane];
  float* base = agg + (size_t)dst * DH + lane * 4;
  atomicAdd(base + 0, xlv.x * alpha);
  atomicAdd(base + 1, xlv.y * alpha);
  atomicAdd(base + 2, xlv.z * alpha);
  atomicAdd(base + 3, xlv.w * alpha);
}

// ------------- GraphNorm -------------
__global__ void k_gn_sum(float* __restrict__ x, const float* __restrict__ bias,
                         const int* __restrict__ bid, float* __restrict__ gsum) {
  long idx = (long)blockIdx.x * 256 + threadIdx.x;
  if (idx >= (long)NNODES * DH) return;
  int n = (int)(idx >> 7), d = (int)(idx & 127);
  float v = x[idx] + bias[d];
  x[idx] = v;
  atomicAdd(&gsum[(size_t)bid[n] * DH + d], v);
}
__global__ void k_gn_mean(float* gsum, const float* __restrict__ cnt) {
  int idx = blockIdx.x * 256 + threadIdx.x;
  if (idx < NGRAPH * DH) gsum[idx] /= cnt[idx >> 7];
}
__global__ void k_gn_center(float* __restrict__ x, const float* __restrict__ mean,
                            const float* __restrict__ ms, const int* __restrict__ bid,
                            float* __restrict__ gvar) {
  long idx = (long)blockIdx.x * 256 + threadIdx.x;
  if (idx >= (long)NNODES * DH) return;
  int n = (int)(idx >> 7), d = (int)(idx & 127);
  float o = x[idx] - mean[(size_t)bid[n] * DH + d] * ms[d];
  x[idx] = o;
  atomicAdd(&gvar[(size_t)bid[n] * DH + d], o * o);
}
__global__ void k_gn_final(const float* __restrict__ x, const float* __restrict__ gvar,
                           const float* __restrict__ cnt, const float* __restrict__ w,
                           const float* __restrict__ b, const int* __restrict__ bid,
                           const float* __restrict__ resid, float* __restrict__ out) {
  long idx = (long)blockIdx.x * 256 + threadIdx.x;
  if (idx >= (long)NNODES * DH) return;
  int n = (int)(idx >> 7), d = (int)(idx & 127);
  int g = bid[n];
  float var = gvar[(size_t)g * DH + d] / cnt[g];
  float r = w[d] * x[idx] * rsqrtf(var + EPSN) + b[d];
  if (resid) r += resid[idx];
  out[idx] = (r > 0.0f) ? r : (expf(r) - 1.0f);  // ELU
}

// ------------- LSTM gates (h0=c0=0): c=sig(i)*tanh(g), h=sig(o)*tanh(c) -------------
__global__ void k_lstm_gate(const float* __restrict__ g, float* __restrict__ h, int rows) {
  long idx = (long)blockIdx.x * 256 + threadIdx.x;
  if (idx >= (long)rows * LSTMH) return;
  int n = (int)(idx >> 8), j = (int)(idx & 255);
  const float* gr = g + (size_t)n * 1024;
  float c = sigm(gr[j]) * tanhf(gr[512 + j]);
  h[(size_t)n * LSTMH + j] = sigm(gr[768 + j]) * tanhf(c);
}

// ------------- LayerNorm over 256 (wave32 per node, 8 vals/lane) -------------
__global__ __launch_bounds__(256)
void k_layernorm(float* __restrict__ h, const float* __restrict__ w,
                 const float* __restrict__ b) {
  int wave = threadIdx.x >> 5, lane = threadIdx.x & 31;
  long n = (long)blockIdx.x * 8 + wave;
  if (n >= NNODES) return;
  float* row = h + n * LSTMH;
  float v[8];
#pragma unroll
  for (int i = 0; i < 8; ++i) v[i] = row[lane * 8 + i];
  float s = 0.0f;
#pragma unroll
  for (int i = 0; i < 8; ++i) s += v[i];
  for (int m = 16; m >= 1; m >>= 1) s += __shfl_xor(s, m, 32);
  float mean = s * (1.0f / 256.0f), q = 0.0f;
#pragma unroll
  for (int i = 0; i < 8; ++i) { float d = v[i] - mean; q += d * d; }
  for (int m = 16; m >= 1; m >>= 1) q += __shfl_xor(q, m, 32);
  float inv = rsqrtf(q * (1.0f / 256.0f) + EPSN);
#pragma unroll
  for (int i = 0; i < 8; ++i) {
    int j = lane * 8 + i;
    row[j] = w[j] * (v[i] - mean) * inv + b[j];
  }
}

__global__ void k_pool(const float* __restrict__ h, const int* __restrict__ bid,
                       float* __restrict__ pooled) {
  long idx = (long)blockIdx.x * 256 + threadIdx.x;
  if (idx >= (long)NNODES * LSTMH) return;
  int n = (int)(idx >> 8), j = (int)(idx & 255);
  atomicAdd(&pooled[(size_t)bid[n] * LSTMH + j], h[idx]);
}

__global__ void k_fc(const float* __restrict__ pooled, const float* __restrict__ cnt,
                     const float* __restrict__ W, const float* __restrict__ b,
                     float* __restrict__ out) {
  int g = threadIdx.x;
  if (g >= NGRAPH) return;
  float inv = 1.0f / cnt[g];
  float lg[4] = {b[0], b[1], b[2], b[3]};
  for (int j = 0; j < LSTMH; ++j) {
    float p = pooled[(size_t)g * LSTMH + j] * inv;
    lg[0] += p * W[j * 4 + 0];
    lg[1] += p * W[j * 4 + 1];
    lg[2] += p * W[j * 4 + 2];
    lg[3] += p * W[j * 4 + 3];
  }
  float mx = fmaxf(fmaxf(lg[0], lg[1]), fmaxf(lg[2], lg[3]));
  float se = 0.0f;
  for (int o = 0; o < 4; ++o) se += expf(lg[o] - mx);
  float lse = mx + logf(se);
  for (int o = 0; o < 4; ++o) out[g * 4 + o] = lg[o] - lse;
}

// ------------- host-side GAT layer driver -------------
static void gat_layer(const float* xin, _Float16* APK,
                      const _Float16* BpWl, const _Float16* BpWr, const _Float16* BpRes,
                      const float* bl, const float* br, const float* att,
                      const float* bias, const float* gnw, const float* gnb,
                      const float* gnms, const float* resb,
                      float* xl, float* xr, float* agg, float* resbuf,
                      float* score, unsigned* smax, float* denom,
                      float* gsum, float* gvar, const float* cnt,
                      const int* eidx, const int* bid, float* xout,
                      hipStream_t s) {
  long apTot = (long)(NNODES / 16) * 4 * 512;  // KT=4 (K=128)
  k_packA<<<(apTot + 255) / 256, 256, 0, s>>>(xin, APK, NNODES, DH);
  dim3 gg((NNODES / 16 + 3) / 4, DH / 64);
  k_gemm_wmma<<<gg, 128, 0, s>>>(APK, BpWl, bl, nullptr, xl, DH, NNODES, 4, 0, 0);
  k_gemm_wmma<<<gg, 128, 0, s>>>(APK, BpWr, br, nullptr, xr, DH, NNODES, 4, 0, 0);
  k_zero<<<(NNODES * 4 + 255) / 256, 256, 0, s>>>((float*)smax, (long)NNODES * 4);
  k_zero<<<(NNODES * 4 + 255) / 256, 256, 0, s>>>(denom, (long)NNODES * 4);
  k_zero<<<(NNODES * DH + 255) / 256, 256, 0, s>>>(agg, (long)NNODES * DH);
  k_score<<<(ETOT + 7) / 8, 256, 0, s>>>(xl, xr, eidx, att, score, smax);
  k_expdenom<<<((long)ETOT * 4 + 255) / 256, 256, 0, s>>>(eidx, smax, score, denom);
  k_aggregate<<<(ETOT + 7) / 8, 256, 0, s>>>(xl, eidx, score, denom, agg);
  const float* resid = nullptr;
  if (BpRes) {
    k_gemm_wmma<<<gg, 128, 0, s>>>(APK, BpRes, resb, nullptr, resbuf, DH, NNODES, 4, 0, 0);
    resid = resbuf;
  }
  k_zero<<<(NGRAPH * DH + 255) / 256, 256, 0, s>>>(gsum, NGRAPH * DH);
  k_zero<<<(NGRAPH * DH + 255) / 256, 256, 0, s>>>(gvar, NGRAPH * DH);
  long nd = (long)NNODES * DH;
  k_gn_sum<<<(nd + 255) / 256, 256, 0, s>>>(agg, bias, bid, gsum);
  k_gn_mean<<<(NGRAPH * DH + 255) / 256, 256, 0, s>>>(gsum, cnt);
  k_gn_center<<<(nd + 255) / 256, 256, 0, s>>>(agg, gsum, gnms, bid, gvar);
  k_gn_final<<<(nd + 255) / 256, 256, 0, s>>>(agg, gvar, cnt, gnw, gnb, bid, resid, xout);
}

extern "C" void kernel_launch(void* const* d_in, const int* in_sizes, int n_in,
                              void* d_out, int out_size, void* d_ws, size_t ws_size,
                              hipStream_t stream) {
  const int*   lab   = (const int*)d_in[0];
  const int*   typ   = (const int*)d_in[1];
  const float* nf    = (const float*)d_in[2];
  const int*   eidx  = (const int*)d_in[3];
  const int*   bid   = (const int*)d_in[4];
  const float* embL  = (const float*)d_in[5];
  const float* embT  = (const float*)d_in[6];
  const float* W_in  = (const float*)d_in[7];
  const float* b_in  = (const float*)d_in[8];
  const float* g1Wl  = (const float*)d_in[9],  *g1bl = (const float*)d_in[10];
  const float* g1Wr  = (const float*)d_in[11], *g1br = (const float*)d_in[12];
  const float* g1att = (const float*)d_in[13], *g1bias = (const float*)d_in[14];
  const float* gn1w  = (const float*)d_in[15], *gn1b = (const float*)d_in[16];
  const float* gn1ms = (const float*)d_in[17];
  const float* g2Wl  = (const float*)d_in[18], *g2bl = (const float*)d_in[19];
  const float* g2Wr  = (const float*)d_in[20], *g2br = (const float*)d_in[21];
  const float* g2att = (const float*)d_in[22], *g2bias = (const float*)d_in[23];
  const float* gn2w  = (const float*)d_in[24], *gn2b = (const float*)d_in[25];
  const float* gn2ms = (const float*)d_in[26];
  const float* Wres  = (const float*)d_in[27], *bres = (const float*)d_in[28];
  const float* l1Wih = (const float*)d_in[29];
  const float* l1bih = (const float*)d_in[31], *l1bhh = (const float*)d_in[32];
  const float* l2Wih = (const float*)d_in[33];
  const float* l2bih = (const float*)d_in[35], *l2bhh = (const float*)d_in[36];
  const float* lnw   = (const float*)d_in[37], *lnb = (const float*)d_in[38];
  const float* fcW   = (const float*)d_in[39], *fcb = (const float*)d_in[40];
  float* out = (float*)d_out;

  // workspace arena (floats, 256B-aligned chunks)
  float* wsF = (float*)d_ws;
  size_t off = 0;
  auto alloc = [&](size_t nfl) { float* p = wsF + off; off += (nfl + 63) & ~(size_t)63; return p; };
  float* X    = alloc((size_t)NNODES * DH);
  float* XL   = alloc((size_t)NNODES * DH);
  float* XR   = alloc((size_t)NNODES * DH);
  float* AGG  = alloc((size_t)NNODES * DH);
  float* X1   = alloc((size_t)NNODES * DH);
  float* SC   = alloc((size_t)ETOT * 4);
  unsigned* SMAX = (unsigned*)alloc((size_t)NNODES * 4);
  float* DEN  = alloc((size_t)NNODES * 4);
  float* GS   = alloc(NGRAPH * DH);
  float* GV   = alloc(NGRAPH * DH);
  float* CNT  = alloc(64);
  float* POOL = alloc(NGRAPH * LSTMH);
  float* H1   = alloc((size_t)NNODES * LSTMH);
  float* H2   = alloc((size_t)NNODES * LSTMH);
  float* GB   = alloc((size_t)LCHUNK * 1024);
  // packed f16 operand buffers (sizes in halves -> floats/2)
  _Float16* APK  = (_Float16*)alloc((size_t)(NNODES / 16) * 8 * 512 / 2);  // up to K=256
  _Float16* BpWl1 = (_Float16*)alloc(DH * DH / 2);
  _Float16* BpWr1 = (_Float16*)alloc(DH * DH / 2);
  _Float16* BpWl2 = (_Float16*)alloc(DH * DH / 2);
  _Float16* BpWr2 = (_Float16*)alloc(DH * DH / 2);
  _Float16* BpRes = (_Float16*)alloc(DH * DH / 2);
  _Float16* BpL1  = (_Float16*)alloc(1024 * DH / 2);
  _Float16* BpL2  = (_Float16*)alloc(1024 * LSTMH / 2);
  (void)ws_size; (void)n_in; (void)in_sizes; (void)out_size;

  // one-time weight packing (fragment order, f16)
  k_packB<<<(DH * DH + 255) / 256, 256, 0, stream>>>(g1Wl, DH, 1, BpWl1, DH, DH);
  k_packB<<<(DH * DH + 255) / 256, 256, 0, stream>>>(g1Wr, DH, 1, BpWr1, DH, DH);
  k_packB<<<(DH * DH + 255) / 256, 256, 0, stream>>>(g2Wl, DH, 1, BpWl2, DH, DH);
  k_packB<<<(DH * DH + 255) / 256, 256, 0, stream>>>(g2Wr, DH, 1, BpWr2, DH, DH);
  k_packB<<<(DH * DH + 255) / 256, 256, 0, stream>>>(Wres, DH, 1, BpRes, DH, DH);
  k_packB<<<(1024 * DH + 255) / 256, 256, 0, stream>>>(l1Wih, 1, DH, BpL1, DH, 1024);
  k_packB<<<(1024 * LSTMH + 255) / 256, 256, 0, stream>>>(l2Wih, 1, LSTMH, BpL2, LSTMH, 1024);

  // graph node counts
  k_zero<<<1, 256, 0, stream>>>(CNT, 64);
  k_count<<<(NNODES + 255) / 256, 256, 0, stream>>>(bid, CNT);

  // input projection
  k_inproj<<<NNODES, 128, 0, stream>>>(lab, typ, nf, embL, embT, W_in, b_in, X);

  // GAT layer 1 -> X1
  gat_layer(X, APK, BpWl1, BpWr1, nullptr, g1bl, g1br, g1att, g1bias,
            gn1w, gn1b, gn1ms, nullptr, XL, XR, AGG, nullptr,
            SC, SMAX, DEN, GS, GV, CNT, eidx, bid, X1, stream);
  // GAT layer 2 (+ residual X1@W_res+b_res) -> XR (=x2)
  gat_layer(X1, APK, BpWl2, BpWr2, BpRes, g2bl, g2br, g2att, g2bias,
            gn2w, gn2b, gn2ms, bres, XL, XR, AGG, XL,
            SC, SMAX, DEN, GS, GV, CNT, eidx, bid, XR, stream);
  float* X2 = XR;

  // LSTM1: g = x2 @ l1_Wih^T + bih + bhh ; gates -> H1   (chunked over rows)
  {
    long apTot = (long)(NNODES / 16) * 4 * 512;
    k_packA<<<(apTot + 255) / 256, 256, 0, stream>>>(X2, APK, NNODES, DH);
    for (int c0 = 0; c0 < NNODES; c0 += LCHUNK) {
      int rows = (NNODES - c0 < LCHUNK) ? (NNODES - c0) : LCHUNK;
      int tiles = (rows + 15) / 16;
      dim3 gg((tiles + 3) / 4, 1024 / 64);
      k_gemm_wmma<<<gg, 128, 0, stream>>>(APK, BpL1, l1bih, l1bhh, GB, 1024,
                                          NNODES, 4, c0 / 16, c0);
      k_lstm_gate<<<((long)rows * LSTMH + 255) / 256, 256, 0, stream>>>(
          GB, H1 + (size_t)c0 * LSTMH, rows);
    }
  }
  // LSTM2: g = h1 @ l2_Wih^T + bih + bhh ; gates -> H2
  {
    long apTot = (long)(NNODES / 16) * 8 * 512;
    k_packA<<<(apTot + 255) / 256, 256, 0, stream>>>(H1, APK, NNODES, LSTMH);
    for (int c0 = 0; c0 < NNODES; c0 += LCHUNK) {
      int rows = (NNODES - c0 < LCHUNK) ? (NNODES - c0) : LCHUNK;
      int tiles = (rows + 15) / 16;
      dim3 gg((tiles + 3) / 4, 1024 / 64);
      k_gemm_wmma<<<gg, 128, 0, stream>>>(APK, BpL2, l2bih, l2bhh, GB, 1024,
                                          NNODES, 8, c0 / 16, c0);
      k_lstm_gate<<<((long)rows * LSTMH + 255) / 256, 256, 0, stream>>>(
          GB, H2 + (size_t)c0 * LSTMH, rows);
    }
  }

  // LayerNorm (in place on H2), mean-pool, FC + log_softmax
  k_layernorm<<<(NNODES + 7) / 8, 256, 0, stream>>>(H2, lnw, lnb);
  k_zero<<<(NGRAPH * LSTMH + 255) / 256, 256, 0, stream>>>(POOL, NGRAPH * LSTMH);
  k_pool<<<((long)NNODES * LSTMH + 255) / 256, 256, 0, stream>>>(H2, bid, POOL);
  k_fc<<<1, 64, 0, stream>>>(POOL, CNT, fcW, fcb, out);
}
